// GVAE_57621281243564
// MI455X (gfx1250) — matrix operand order repaired and checked by reading, not compiled, not run on hardware
//
#include <hip/hip_runtime.h>
#include <hip/hip_bf16.h>

typedef __attribute__((ext_vector_type(16))) _Float16 v16h;
typedef __attribute__((ext_vector_type(8)))  _Float16 h8;
typedef __attribute__((ext_vector_type(8)))  float    v8f;

__device__ __forceinline__ h8 cvt8(float4 a, float4 b) {
    h8 r;
    r[0] = (_Float16)a.x; r[1] = (_Float16)a.y; r[2] = (_Float16)a.z; r[3] = (_Float16)a.w;
    r[4] = (_Float16)b.x; r[5] = (_Float16)b.y; r[6] = (_Float16)b.z; r[7] = (_Float16)b.w;
    return r;
}

// ---------------------------------------------------------------------------
// Generic GEMM: C[M,N] = act(A[M,K] @ W[K,N] + bias[N]),  act: 0=none, 1=relu
// 128 threads = 4 waves; block computes a 32x32 C tile, each wave one 16x16
// quadrant via v_wmma_f32_16x16x32_f16 (f16 inputs, f32 accumulate).
//
// FULL=true  : M,K,N all multiples of 32 -> unchecked float4 global loads,
//              packed h8 (ds_store_b128) staging, b128 fragment builds.
// FULL=false : ragged edges (the tiny 12/13-dim layers) -> checked loads.
//
// LDS layout: As[m][k] row-major; Bs staged TRANSPOSED as [n][k] so each
// lane's B-fragment halves (fixed n, contiguous k) are two ds_load_b128.
// Row stride 40 halves = 80 bytes keeps 16-byte alignment.
// ---------------------------------------------------------------------------
template <bool FULL>
__global__ __launch_bounds__(128)
void gemm_kern(const float* __restrict__ A, const float* __restrict__ Bw,
               const float* __restrict__ bias, float* __restrict__ C,
               int M, int K, int N, int act)
{
    __shared__ alignas(16) _Float16 As[32][40];   // [m][k]
    __shared__ alignas(16) _Float16 Bs[32][40];   // [n][k]  (transposed stage)
    const int tid  = threadIdx.x;
    const int wave = tid >> 5;
    const int lane = tid & 31;
    const int wy   = wave >> 1;        // M quadrant (0..1)
    const int wx   = wave & 1;         // N quadrant (0..1)
    const int hi   = lane >> 4;        // half-wave selector
    const int tileM = blockIdx.y << 5;
    const int tileN = blockIdx.x << 5;

    v8f acc = {};

    for (int k0 = 0; k0 < K; k0 += 32) {
        if (FULL) {
            // 8 elems per thread for each tile: 2x global_load_b128 + cvt + b128 LDS store
            const int r  = tid >> 2;             // 0..31
            const int c0 = (tid & 3) << 3;       // 0,8,16,24
            const float* ap = A + (size_t)(tileM + r) * K + (k0 + c0);
            float4 a0 = *(const float4*)ap;
            float4 a1 = *(const float4*)(ap + 4);
            *(h8*)&As[r][c0] = cvt8(a0, a1);

            const float* bp = Bw + (size_t)(k0 + r) * N + (tileN + c0);
            float4 b0 = *(const float4*)bp;
            float4 b1 = *(const float4*)(bp + 4);
            h8 bv = cvt8(b0, b1);
            #pragma unroll
            for (int j = 0; j < 8; ++j) Bs[c0 + j][r] = bv[j];   // transpose into LDS

            // next B k-tile is streaming weight traffic -> hint it into cache
            if (k0 + 32 < K) __builtin_prefetch(bp + (size_t)32 * N, 0, 0);
        } else {
            for (int e = tid; e < 1024; e += 128) {
                int r = e >> 5, c = e & 31;
                int ar = tileM + r, ac = k0 + c;
                As[r][c] = (_Float16)((ar < M && ac < K) ? A[(size_t)ar * K + ac] : 0.f);
                int br = k0 + r, bc = tileN + c;
                Bs[c][r] = (_Float16)((br < K && bc < N) ? Bw[(size_t)br * N + bc] : 0.f);
            }
        }
        __syncthreads();

        // Wave32 fragments per CDNA5 ISA 7.12.2:
        // A 16x32 f16: lanes 0-15 row M=lane hold K={0..7,16..23};
        //              lanes 16-31 row M=lane-16 hold K={8..15,24..31}.
        //   -> two contiguous 8-half runs of an As row: 2x ds_load_b128.
        // B 32x16 f16: lanes 0-15 col N=lane hold K={0..7,16..23};
        //              lanes 16-31 col N=lane-16 hold K={8..15,24..31}.
        //   -> two contiguous 8-half runs of a Bs row: 2x ds_load_b128.
        const _Float16* arow = &As[(lane & 15) + (wy << 4)][hi << 3];
        h8 alo = *(const h8*)arow;
        h8 ahi = *(const h8*)(arow + 16);
        v16h af = __builtin_shufflevector(alo, ahi, 0, 1, 2, 3, 4, 5, 6, 7,
                                          8, 9, 10, 11, 12, 13, 14, 15);
        const _Float16* brow = &Bs[(lane & 15) + (wx << 4)][hi << 3];
        h8 blo = *(const h8*)brow;
        h8 bhi = *(const h8*)(brow + 16);
        v16h bf = __builtin_shufflevector(blo, bhi, 0, 1, 2, 3, 4, 5, 6, 7,
                                          8, 9, 10, 11, 12, 13, 14, 15);

        acc = __builtin_amdgcn_wmma_f32_16x16x32_f16(false, af, false, bf,
                                                     (short)0, acc, false, false);
        __syncthreads();
    }

    // D fragment: lane<16 -> rows j (0..7); lane>=16 -> rows 8+j; col = lane&15
    const int ncol = tileN + (wx << 4) + (lane & 15);
    #pragma unroll
    for (int j = 0; j < 8; ++j) {
        int mrow = tileM + (wy << 4) + ((lane < 16) ? j : j + 8);
        if (FULL || (mrow < M && ncol < N)) {
            float v = acc[j] + bias[ncol];
            if (act) v = fmaxf(v, 0.f);
            C[(size_t)mrow * N + ncol] = v;
        }
    }
}

// ---------------------------------------------------------------------------
// Fused MHA core: one block per (b,h,i); 512 threads = 16 waves, wave j lane d.
// attn  = q_i * k_j / sqrt(D)          (elementwise, never hits HBM)
// new_e = em*attn + ea + attn          (materialized: feeds the 'eo' GEMM)
// a     = softmax_j(sum_d new_e)
// wv    = sum_j a_j * v_j
// ---------------------------------------------------------------------------
__global__ __launch_bounds__(512)
void mha_fuse(const float* __restrict__ q, const float* __restrict__ k,
              const float* __restrict__ v, const float* __restrict__ em,
              const float* __restrict__ ea,
              float* __restrict__ ne, float* __restrict__ wv)
{
    const int blk = blockIdx.x;              // b*H*N + h*N + i
    const int i = blk & 15;
    const int h = (blk >> 4) & 7;
    const int b = blk >> 7;
    const int tid = threadIdx.x;
    const int j = tid >> 5;                  // 0..15  (one wave per j)
    const int d = tid & 31;                  // 0..31

    __shared__ float sc[16];
    __shared__ float aw[16];
    const float scale = 0.1767766952966369f; // 1/sqrt(32)

    const size_t qo = ((size_t)(b * 16 + i) * 256) + h * 32 + d;
    const size_t ko = ((size_t)(b * 16 + j) * 256) + h * 32 + d;
    const size_t eo = ((size_t)((b * 16 + i) * 16 + j) * 256) + h * 32 + d;

    float attn = q[qo] * k[ko] * scale;
    float nev  = em[eo] * attn + ea[eo] + attn;
    ne[eo] = nev;

    float s = nev;                            // wave32 reduce over d
    #pragma unroll
    for (int off = 16; off; off >>= 1) s += __shfl_xor(s, off, 32);
    if (d == 0) sc[j] = s;
    __syncthreads();

    if (tid == 0) {                           // softmax over 16 scores
        float mx = sc[0];
        for (int t = 1; t < 16; ++t) mx = fmaxf(mx, sc[t]);
        float sum = 0.f;
        for (int t = 0; t < 16; ++t) { float e = __expf(sc[t] - mx); aw[t] = e; sum += e; }
        float inv = 1.f / sum;
        for (int t = 0; t < 16; ++t) aw[t] *= inv;
    }
    __syncthreads();

    if (tid < 32) {                           // wv[i,h,d] = sum_j a_j * v[j,h,d]
        float acc = 0.f;
        for (int t = 0; t < 16; ++t)
            acc += aw[t] * v[((size_t)(b * 16 + t) * 256) + h * 32 + tid];
        wv[((size_t)(b * 16 + i) * 256) + h * 32 + tid] = acc;
    }
}

// ---------------------------------------------------------------------------
// out = LN(x + res) with mean/var over the whole per-batch slice of L elems
// (reference normalizes over (N,D) / (N,N,D)). gamma/beta indexed by position.
// One block (256 thr) per batch element.
// ---------------------------------------------------------------------------
__global__ __launch_bounds__(256)
void add_ln(const float* __restrict__ x, const float* __restrict__ res,
            const float* __restrict__ g, const float* __restrict__ bta,
            float* __restrict__ out, int L)
{
    const int b = blockIdx.x, tid = threadIdx.x;
    const size_t base = (size_t)b * L;
    float s = 0.f, ss = 0.f;
    for (int idx = tid; idx < L; idx += 256) {
        float hv = x[base + idx] + res[base + idx];
        s += hv; ss += hv * hv;
    }
    #pragma unroll
    for (int off = 16; off; off >>= 1) {
        s  += __shfl_xor(s,  off, 32);
        ss += __shfl_xor(ss, off, 32);
    }
    __shared__ float rs[8], rss[8];
    __shared__ float mean_s, inv_s;
    const int wave = tid >> 5, lane = tid & 31;
    if (lane == 0) { rs[wave] = s; rss[wave] = ss; }
    __syncthreads();
    if (tid == 0) {
        float S = 0.f, SS = 0.f;
        for (int t = 0; t < 8; ++t) { S += rs[t]; SS += rss[t]; }
        float m = S / (float)L;
        float vv = SS / (float)L - m * m;
        mean_s = m; inv_s = rsqrtf(vv + 1e-5f);
    }
    __syncthreads();
    const float m = mean_s, inv = inv_s;
    for (int idx = tid; idx < L; idx += 256) {
        float hv = x[base + idx] + res[base + idx];
        out[base + idx] = (hv - m) * inv * g[idx] + bta[idx];
    }
}

// z = means + eps * exp(0.5 * logvars)
__global__ void reparam_k(const float* __restrict__ m, const float* __restrict__ lv,
                          const float* __restrict__ eps, float* __restrict__ z, int n)
{
    int t = blockIdx.x * blockDim.x + threadIdx.x;
    if (t < n) z[t] = m[t] + eps[t] * __expf(0.5f * lv[t]);
}

// node head: [sigmoid(x0), softmax(x1..x5), x6..x12] per (b,n)
__global__ void node_head(const float* __restrict__ x, float* __restrict__ out, int total)
{
    int t = blockIdx.x * blockDim.x + threadIdx.x;
    if (t >= total) return;
    const float* xi = x + (size_t)t * 13;
    float* o = out + (size_t)t * 13;
    o[0] = 1.f / (1.f + __expf(-xi[0]));
    float mx = xi[1];
    for (int i = 2; i < 6; ++i) mx = fmaxf(mx, xi[i]);
    float e[5], sum = 0.f;
    for (int i = 0; i < 5; ++i) { e[i] = __expf(xi[1 + i] - mx); sum += e[i]; }
    float inv = 1.f / sum;
    for (int i = 0; i < 5; ++i) o[1 + i] = e[i] * inv;
    for (int i = 6; i < 13; ++i) o[i] = xi[i];
}

// edge head: three softmax groups of 4 over the 12 dims, per (b,i,j)
__global__ void edge_head(const float* __restrict__ x, float* __restrict__ out, int total)
{
    int t = blockIdx.x * blockDim.x + threadIdx.x;
    if (t >= total) return;
    const float* xi = x + (size_t)t * 12;
    float* o = out + (size_t)t * 12;
    for (int gdx = 0; gdx < 3; ++gdx) {
        const float* xg = xi + gdx * 4;
        float* og = o + gdx * 4;
        float mx = fmaxf(fmaxf(xg[0], xg[1]), fmaxf(xg[2], xg[3]));
        float e0 = __expf(xg[0] - mx), e1 = __expf(xg[1] - mx);
        float e2 = __expf(xg[2] - mx), e3 = __expf(xg[3] - mx);
        float inv = 1.f / (e0 + e1 + e2 + e3);
        og[0] = e0 * inv; og[1] = e1 * inv; og[2] = e2 * inv; og[3] = e3 * inv;
    }
}

// ---------------------------------------------------------------------------
struct Lin { const float* w; const float* b; };
struct TFLayer {
    Lin q, k, v, mul, add, no, eo;
    const float *l1n_g, *l1n_b, *l1e_g, *l1e_b;
    Lin mn0, mn1, me0, me1;
    const float *l2n_g, *l2n_b, *l2e_g, *l2e_b;
};

extern "C" void kernel_launch(void* const* d_in, const int* in_sizes, int n_in,
                              void* d_out, int out_size, void* d_ws, size_t ws_size,
                              hipStream_t stream)
{
    (void)in_sizes; (void)n_in; (void)out_size; (void)ws_size;

    const float* nodes = (const float*)d_in[0];   // [64,16,13]
    const float* edges = (const float*)d_in[1];   // [64,16,16,12]
    const float* eps   = (const float*)d_in[2];   // [64,512]

    // --- params walked in setup_inputs() insertion order ---
    int p = 3;
    auto F   = [&](int i) { return (const float*)d_in[i]; };
    auto lin = [&]() { Lin L{F(p), F(p + 1)}; p += 2; return L; };
    auto layer = [&]() {
        TFLayer L;
        L.q = lin(); L.k = lin(); L.v = lin(); L.mul = lin(); L.add = lin();
        L.no = lin(); L.eo = lin();
        L.l1n_g = F(p++); L.l1n_b = F(p++); L.l1e_g = F(p++); L.l1e_b = F(p++);
        L.mn0 = lin(); L.mn1 = lin(); L.me0 = lin(); L.me1 = lin();
        L.l2n_g = F(p++); L.l2n_b = F(p++); L.l2e_g = F(p++); L.l2e_b = F(p++);
        return L;
    };

    Lin in_n0 = lin(), in_n1 = lin(), in_e0 = lin(), in_e1 = lin();
    TFLayer encL[8]; for (int i = 0; i < 8; ++i) encL[i] = layer();
    Lin mean0 = lin(), mean1 = lin(), logv0 = lin(), logv1 = lin();
    Lin cn0 = lin(), cn1 = lin(), ce0 = lin(), ce1 = lin();
    TFLayer decL[8]; for (int i = 0; i < 8; ++i) decL[i] = layer();
    Lin on0 = lin(), on1 = lin(), oe0 = lin(), oe1 = lin();

    // --- workspace carve (floats) ---
    float* w = (float*)d_ws;
    auto alloc = [&](size_t n) { float* r = w; w += n; return r; };
    float* bufN   = alloc(32768);    // node activations  [B*N, 32]
    float* bufE   = alloc(524288);   // edge activations  [B*N*N, 32]
    float* bufQ   = alloc(262144);
    float* bufK   = alloc(262144);
    float* bufV   = alloc(262144);
    float* bufWV  = alloc(262144);
    float* bufEM  = alloc(4194304);
    float* bufEA  = alloc(4194304);
    float* bufNE  = alloc(4194304);
    float* bufAN  = alloc(32768);
    float* bufAN2 = alloc(32768);
    float* bufAE  = alloc(524288);
    float* bufAE2 = alloc(524288);
    float* bufZ   = alloc(32768);
    float* T1     = alloc(1048576);
    float* T2     = alloc(1048576);

    float* outF      = (float*)d_out;
    float* out_n     = outF;             // 64*16*13    = 13312
    float* out_e     = outF + 13312;     // 64*16*16*12 = 196608
    float* out_means = outF + 209920;    // 64*512
    float* out_logv  = outF + 242688;    // 64*512

    auto GEMM = [&](const float* A, Lin L, float* C, int M, int K, int N, int act) {
        dim3 g((N + 31) / 32, (M + 31) / 32);
        if ((M % 32 == 0) && (K % 32 == 0) && (N % 32 == 0))
            gemm_kern<true><<<g, 128, 0, stream>>>(A, L.w, L.b, C, M, K, N, act);
        else
            gemm_kern<false><<<g, 128, 0, stream>>>(A, L.w, L.b, C, M, K, N, act);
    };

    auto tf = [&](const TFLayer& L) {
        GEMM(bufN, L.q,   bufQ,  1024,  32, 256, 0);
        GEMM(bufN, L.k,   bufK,  1024,  32, 256, 0);
        GEMM(bufN, L.v,   bufV,  1024,  32, 256, 0);
        GEMM(bufE, L.mul, bufEM, 16384, 32, 256, 0);
        GEMM(bufE, L.add, bufEA, 16384, 32, 256, 0);
        mha_fuse<<<8192, 512, 0, stream>>>(bufQ, bufK, bufV, bufEM, bufEA, bufNE, bufWV);
        GEMM(bufWV, L.no, bufAN, 1024,  256, 32, 0);
        GEMM(bufNE, L.eo, bufAE, 16384, 256, 32, 0);
        add_ln<<<64, 256, 0, stream>>>(bufAN, bufN, L.l1n_g, L.l1n_b, bufAN2, 512);
        add_ln<<<64, 256, 0, stream>>>(bufAE, bufE, L.l1e_g, L.l1e_b, bufAE2, 8192);
        GEMM(bufAN2, L.mn0, T1,    1024, 32, 32, 1);
        GEMM(T1,     L.mn1, bufAN, 1024, 32, 32, 1);
        add_ln<<<64, 256, 0, stream>>>(bufAN, bufAN2, L.l2n_g, L.l2n_b, bufN, 512);
        GEMM(bufAE2, L.me0, T1,    16384, 32, 32, 1);
        GEMM(T1,     L.me1, bufAE, 16384, 32, 32, 1);
        add_ln<<<64, 256, 0, stream>>>(bufAE, bufAE2, L.l2e_g, L.l2e_b, bufE, 8192);
    };

    // ---- encoder input MLPs ----
    GEMM(nodes, in_n0, T1,   1024,  13, 64, 1);
    GEMM(T1,    in_n1, bufN, 1024,  64, 32, 1);
    GEMM(edges, in_e0, T2,   16384, 12, 64, 1);
    GEMM(T2,    in_e1, bufE, 16384, 64, 32, 1);

    // ---- encoder transformer stack ----
    for (int l = 0; l < 8; ++l) tf(encL[l]);

    // ---- latent: g = bufN viewed as [64, 512] (same memory layout) ----
    GEMM(bufN, mean0, T1,        64, 512, 512, 1);
    GEMM(T1,   mean1, out_means, 64, 512, 512, 1);
    GEMM(bufN, logv0, T1,        64, 512, 512, 1);
    GEMM(T1,   logv1, out_logv,  64, 512, 512, 1);
    reparam_k<<<(32768 + 255) / 256, 256, 0, stream>>>(out_means, out_logv, eps, bufZ, 32768);

    // ---- decoder conditioning (reshapes are layout no-ops) ----
    GEMM(bufZ, cn0, T1,   64, 512,  512,  1);
    GEMM(T1,   cn1, bufN, 64, 512,  512,  1);
    GEMM(bufZ, ce0, T1,   64, 512,  8192, 1);
    GEMM(T1,   ce1, bufE, 64, 8192, 8192, 1);   // HBM-bound: streams 256MB of weights

    // ---- decoder transformer stack ----
    for (int l = 0; l < 8; ++l) tf(decL[l]);

    // ---- output heads ----
    GEMM(bufN, on0, T1, 1024, 32, 13, 1);
    GEMM(T1,   on1, T2, 1024, 13, 13, 0);
    node_head<<<(1024 + 255) / 256, 256, 0, stream>>>(T2, out_n, 1024);

    GEMM(bufE, oe0, T1, 16384, 32, 12, 1);
    GEMM(T1,   oe1, T2, 16384, 12, 12, 0);
    edge_head<<<(16384 + 255) / 256, 256, 0, stream>>>(T2, out_e, 16384);
}
